// BiGRUWithAttention_88613765251271
// MI455X (gfx1250) — matrix operand order repaired and checked by reading, not compiled, not run on hardware
//
#include <hip/hip_runtime.h>
#include <hip/hip_bf16.h>
#include <math.h>
#include <stdint.h>

typedef __bf16 bf16;
typedef __attribute__((ext_vector_type(16))) __bf16 v16bf;
typedef __attribute__((ext_vector_type(8)))  float  v8f;
typedef __attribute__((ext_vector_type(4)))  float  f32x4;
typedef __attribute__((ext_vector_type(4)))  unsigned int u32x4;
typedef __attribute__((ext_vector_type(8)))  int  i32x8;
typedef __attribute__((ext_vector_type(4)))  int  i32x4;

union FragU { v16bf v; struct { f32x4 lo, hi; } q; };

#if defined(__HIP_DEVICE_COMPILE__)
#if __has_builtin(__builtin_amdgcn_tensor_load_to_lds)
#define HAVE_TDM_BUILTIN 1
#pragma message("ATHENA-INFO: tensor_load_to_lds BUILTIN path")
#else
#define HAVE_TDM_BUILTIN 0
#pragma message("ATHENA-INFO: tensor_load_to_lds INLINE-ASM path")
#endif
#else
#define HAVE_TDM_BUILTIN 0
#endif

typedef __attribute__((address_space(3))) bf16 as3_bf16;
__device__ __forceinline__ unsigned lds_addr_of(bf16* p) {
  // generic -> LDS addrspace cast, then ptrtoint gives the LDS byte offset
  return (unsigned)(uintptr_t)(as3_bf16*)p;
}

__device__ __forceinline__ void wait_tensorcnt0() {
#if defined(__HIP_DEVICE_COMPILE__) && __has_builtin(__builtin_amdgcn_s_wait_tensorcnt)
  __builtin_amdgcn_s_wait_tensorcnt((short)0);
#else
  asm volatile("s_wait_tensorcnt 0x0" ::: "memory");
#endif
}
__device__ __forceinline__ void wait_tensorcnt1() {
#if defined(__HIP_DEVICE_COMPILE__) && __has_builtin(__builtin_amdgcn_s_wait_tensorcnt)
  __builtin_amdgcn_s_wait_tensorcnt((short)1);
#else
  asm volatile("s_wait_tensorcnt 0x1" ::: "memory");
#endif
}

// ---------------------------------------------------------------------------
// TDM: DMA a 64-row x 64-col bf16 weight panel (tile of W[N,K], row-major)
// into LDS at lds_byte_addr. tensor_dim1 = rowsRem so OOB rows read as zero.
// D# per CDNA5 ISA §8.3/8.4: group0 {count=1, lds_addr, global_addr, type=2};
// group1 {data_size=2B, tensor_dim0=K, tensor_dim1=rowsRem, tile_dim0=64,
// tile_dim1=64, tensor_dim0_stride=K}. Tracked by TENSORcnt.
// ---------------------------------------------------------------------------
__device__ __forceinline__ void tdm_issue(const bf16* W, int nblk, int K,
                                          int rowsRem, int kk,
                                          unsigned lds_byte_addr) {
  const bf16* src = W + (size_t)nblk * K + kk;
  unsigned long long ga = (unsigned long long)(uintptr_t)src;
  u32x4 g0;
  g0[0] = 1u;                                        // count=1 (valid user D#)
  g0[1] = lds_byte_addr;                             // lds_addr [63:32]
  g0[2] = (unsigned)(ga & 0xffffffffu);              // global_addr lo
  g0[3] = (unsigned)((ga >> 32) & 0x01ffffffu) | (2u << 30);  // addr hi+type=2
  i32x8 g1;
  g1[0] = 0x00010000;                                // data_size=1 (2 bytes)
  g1[1] = (int)(((unsigned)K & 0xffffu) << 16);      // tensor_dim0 lo16
  g1[2] = (int)((((unsigned)K >> 16) & 0xffffu) |
                (((unsigned)rowsRem & 0xffffu) << 16));  // dim0 hi | dim1 lo
  g1[3] = (int)((((unsigned)rowsRem >> 16) & 0xffffu) |
                (64u << 16));                        // dim1 hi | tile_dim0=64
  g1[4] = 64;                                        // tile_dim1=64, tile_dim2=0
  g1[5] = K;                                         // tensor_dim0_stride lo32
  g1[6] = 0;
  g1[7] = 0;
  i32x4 g2 = {0, 0, 0, 0};
  i32x4 g3 = {0, 0, 0, 0};
#if HAVE_TDM_BUILTIN
#if defined(__clang_major__) && __clang_major__ >= 23
  i32x8 g4 = {0, 0, 0, 0, 0, 0, 0, 0};
  __builtin_amdgcn_tensor_load_to_lds(g0, g1, g2, g3, g4, 0);
#else
  __builtin_amdgcn_tensor_load_to_lds(g0, g1, g2, g3, 0);
#endif
#else
  asm volatile("tensor_load_to_lds %0, %1, %2, %3"
               :
               : "s"(g0), "s"(g1), "s"(g2), "s"(g3)
               : "memory");
#endif
}

// ---------------------------------------------------------------------------
// A-fragment (16x32 bf16, row-major source, ld in elements).
// Lanes 0-15: row M=lane, K = k0+0..7 (v0..7) and k0+16..23 (v8..15);
// lanes 16-31: K = k0+8..15 and k0+24..31.
// ---------------------------------------------------------------------------
__device__ __forceinline__ v16bf load_frag_a(const bf16* __restrict__ p, int ld,
                                             int row0, int k0, int maxRow) {
  int lane = threadIdx.x & 31;
  int r = row0 + (lane & 15);
  if (r > maxRow) r = maxRow;
  int kx = k0 + ((lane & 16) >> 1);
  const bf16* qp = p + (size_t)r * ld + kx;
  FragU u;
  u.q.lo = *(const f32x4*)(qp);
  u.q.hi = *(const f32x4*)(qp + 16);
  return u.v;
}

// ---------------------------------------------------------------------------
// B-fragment (32x16 bf16) from W[N,K] row-major: lane column n = lane&15,
// lanes 0-15 hold K=k0+0..7 and k0+16..23, lanes 16-31 hold K=k0+8..15 and k0+24..31.
// ---------------------------------------------------------------------------
__device__ __forceinline__ v16bf load_frag_b(const bf16* __restrict__ w, int ld,
                                             int n0, int k0, int maxRow) {
  int lane = threadIdx.x & 31;
  int r = n0 + (lane & 15);
  if (r > maxRow) r = maxRow;
  int kx = k0 + ((lane & 16) >> 1);
  const bf16* qp = w + (size_t)r * ld + kx;
  FragU u;
  u.q.lo = *(const f32x4*)(qp);
  u.q.hi = *(const f32x4*)(qp + 16);
  return u.v;
}

// ---------------------------------------------------------------------------
// fp32 -> bf16 conversion
// ---------------------------------------------------------------------------
__global__ __launch_bounds__(256) void f32_to_bf16_kernel(
    const float* __restrict__ in, bf16* __restrict__ out, int n) {
  int i = blockIdx.x * 256 + threadIdx.x;
  if (i < n) out[i] = (bf16)in[i];
}

// ---------------------------------------------------------------------------
// WMMA GEMM: C[M,N] = A[M,K] * W[N,K]^T (+bias).
// 256 threads = 8 waves; block tile 32(M) x 64(N); one 16x16 tile per wave.
// Weight panels (64 rows x 64 K, 8 KB) staged in LDS by the Tensor Data
// Mover, double buffered: DMA of panel i+1 overlaps the two WMMAs on panel i
// (TENSORcnt pipelining; workgroup barrier publishes each panel).
// EPI 0: Cf = acc + bias. EPI 1: Cb = bf16(tanh(acc + bias) * Gate).
// ---------------------------------------------------------------------------
template <int EPI>
__global__ __launch_bounds__(256) void gemm_wmma_bf16(
    const bf16* __restrict__ A, const bf16* __restrict__ W,
    const float* __restrict__ bias, float* __restrict__ Cf,
    bf16* __restrict__ Cb, const float* __restrict__ Gate,
    int M, int N, int K, int ldc) {
  __shared__ __align__(128) bf16 ldsW[2][64 * 64];
  int lane = threadIdx.x & 31;
  int wave = threadIdx.x >> 5;
  int m0 = blockIdx.y * 32 + (wave >> 2) * 16;
  int nblk = blockIdx.x * 64;
  int nloc = (wave & 3) * 16;
  int nk = K / 64;
  int rowsRem = N - nblk;

  unsigned lds0 = lds_addr_of(&ldsW[0][0]);
  unsigned lds1 = lds_addr_of(&ldsW[1][0]);
  if (wave == 0) tdm_issue(W, nblk, K, rowsRem, 0, lds0);

  v8f acc = {0.f, 0.f, 0.f, 0.f, 0.f, 0.f, 0.f, 0.f};
  for (int i = 0; i < nk; i++) {
    int kk = i * 64;
    if (wave == 0) {
      if (i + 1 < nk) {
        tdm_issue(W, nblk, K, rowsRem, kk + 64, ((i + 1) & 1) ? lds1 : lds0);
        wait_tensorcnt1();
      } else {
        wait_tensorcnt0();
      }
    }
    __syncthreads();  // panel i visible to all waves

    const bf16* wb = &ldsW[i & 1][0];
    int nb = nloc + (lane & 15);
    if (kk + 128 < K) {  // prefetch A two panels ahead
      __builtin_prefetch(A + (size_t)(m0 + (lane & 15)) * K + kk + 128, 0, 1);
    }
#pragma unroll
    for (int h = 0; h < 2; h++) {
      v16bf a = load_frag_a(A, K, m0, kk + h * 32, M - 1);
      const bf16* qp = &wb[nb * 64 + h * 32 + ((lane & 16) >> 1)];
      FragU ub;
      ub.q.lo = *(const f32x4*)(qp);
      ub.q.hi = *(const f32x4*)(qp + 16);
      acc = __builtin_amdgcn_wmma_f32_16x16x32_bf16(false, a, false, ub.v,
                                                    (short)0, acc, false,
                                                    false);
    }
    __syncthreads();  // all reads of this buffer done before it is reused
  }

  // C/D layout: lane l, vgpr r -> (m = r + 8*(l>>4), n = l&15)
  int n = nblk + nloc + (lane & 15);
  int mb = m0 + ((lane >> 4) << 3);
  if (n < N) {
    float bv = bias ? bias[n] : 0.f;
#pragma unroll
    for (int r = 0; r < 8; r++) {
      size_t idx = (size_t)(mb + r) * (size_t)ldc + (size_t)n;
      float v = acc[r] + bv;
      if (EPI == 0) {
        Cf[idx] = v;
      } else {
        v = tanhf(v) * Gate[idx];
        Cb[idx] = (bf16)v;
      }
    }
  }
}

// ---------------------------------------------------------------------------
// Persistent GRU direction/batch-tile workgroup (lane-local gate math, fp32
// hidden state in registers, bf16 h in LDS for the next step's WMMA
// A-fragments; two s_barriers per timestep).
// grid = 4: blockIdx = dir*2 + batch_half. 1024 threads = 32 waves.
// ---------------------------------------------------------------------------
__global__ __launch_bounds__(1024) void gru_layer_kernel(
    const float* __restrict__ xg,   // [2][32768][1536] rows b*T+t, bih folded
    const bf16* __restrict__ Whh,   // [2][1536][512] bf16
    const float* __restrict__ bhh,  // [2][1536] fp32
    bf16* __restrict__ outb,        // [32768][1024] bf16, cols dir*512..+512
    float* __restrict__ outf,       // optional fp32 mirror
    int writeF32) {
  __shared__ bf16 hb[16 * 512];

  const int T = 1024;
  int dir = blockIdx.x >> 1;
  int bh = blockIdx.x & 1;
  const float* xgd = xg + (size_t)dir * (32768ull * 1536ull);
  const bf16* Wd = Whh + (size_t)dir * (1536ull * 512ull);
  const float* bd = bhh + dir * 1536;

  int tid = threadIdx.x, lane = tid & 31, wave = tid >> 5;
  int jn = wave * 16 + (lane & 15);
  int mrow = (lane >> 4) << 3;

  float b_r = bd[jn];
  float b_z = bd[512 + jn];
  float b_n = bd[1024 + jn];

  float hold[8];
#pragma unroll
  for (int r = 0; r < 8; r++) hold[r] = 0.f;
  for (int i = tid; i < 16 * 512; i += 1024) hb[i] = (bf16)0.f;
  __syncthreads();

  for (int s = 0; s < T; s++) {
    int t = dir ? (T - 1 - s) : s;

    v8f aR = {0.f, 0.f, 0.f, 0.f, 0.f, 0.f, 0.f, 0.f};
    v8f aZ = {0.f, 0.f, 0.f, 0.f, 0.f, 0.f, 0.f, 0.f};
    v8f aN = {0.f, 0.f, 0.f, 0.f, 0.f, 0.f, 0.f, 0.f};
#pragma unroll 4
    for (int kk = 0; kk < 512; kk += 32) {
      FragU ua;
      {
        int rr = lane & 15;
        int kx = kk + ((lane & 16) >> 1);
        const bf16* qp = &hb[rr * 512 + kx];
        ua.q.lo = *(const f32x4*)(qp);
        ua.q.hi = *(const f32x4*)(qp + 16);
      }
      v16bf bR = load_frag_b(Wd, 512, wave * 16, kk, 1535);
      v16bf bZ = load_frag_b(Wd, 512, 512 + wave * 16, kk, 1535);
      v16bf bN = load_frag_b(Wd, 512, 1024 + wave * 16, kk, 1535);
      aR = __builtin_amdgcn_wmma_f32_16x16x32_bf16(false, ua.v, false, bR,
                                                   (short)0, aR, false, false);
      aZ = __builtin_amdgcn_wmma_f32_16x16x32_bf16(false, ua.v, false, bZ,
                                                   (short)0, aZ, false, false);
      aN = __builtin_amdgcn_wmma_f32_16x16x32_bf16(false, ua.v, false, bN,
                                                   (short)0, aN, false, false);
    }
    __syncthreads();

#pragma unroll
    for (int r = 0; r < 8; r++) {
      int m = mrow + r;
      int b = bh * 16 + m;
      size_t xoff = ((size_t)(b * 1024 + t)) * 1536;
      float xr = xgd[xoff + jn];
      float xz = xgd[xoff + 512 + jn];
      float xn = xgd[xoff + 1024 + jn];
      float rg = 1.f / (1.f + __expf(-(xr + aR[r] + b_r)));
      float zg = 1.f / (1.f + __expf(-(xz + aZ[r] + b_z)));
      float ng = tanhf(xn + rg * (aN[r] + b_n));
      float hn = (1.f - zg) * ng + zg * hold[r];
      hold[r] = hn;
      hb[m * 512 + jn] = (bf16)hn;
      size_t orow = ((size_t)(b * 1024 + t)) * 1024 + (size_t)dir * 512 + jn;
      outb[orow] = (bf16)hn;
      if (writeF32) outf[orow] = hn;
    }
    __syncthreads();
  }
}

// ---------------------------------------------------------------------------
// Host orchestration
// ---------------------------------------------------------------------------
extern "C" void kernel_launch(void* const* d_in, const int* in_sizes, int n_in,
                              void* d_out, int out_size, void* d_ws,
                              size_t ws_size, hipStream_t stream) {
  const float* x = (const float*)d_in[0];
  const float* Wih0 = (const float*)d_in[1];
  const float* Whh0 = (const float*)d_in[2];
  const float* bih0 = (const float*)d_in[3];
  const float* bhh0 = (const float*)d_in[4];
  const float* Wih1 = (const float*)d_in[5];
  const float* Whh1 = (const float*)d_in[6];
  const float* bih1 = (const float*)d_in[7];
  const float* bhh1 = (const float*)d_in[8];
  const float* attnW = (const float*)d_in[9];
  const float* attnb = (const float*)d_in[10];
  const float* fcW = (const float*)d_in[11];
  const float* fcb = (const float*)d_in[12];
  float* out = (float*)d_out;

  const size_t M = 32768;

  char* ws = (char*)d_ws;
  size_t off = 0;
  auto wsalloc = [&](size_t bytes) -> char* {
    char* p = ws + off;
    off = (off + bytes + 255) & ~(size_t)255;
    return p;
  };
  bf16* Wih0b = (bf16*)wsalloc(2ull * 1536 * 128 * 2);
  bf16* Whh0b = (bf16*)wsalloc(2ull * 1536 * 512 * 2);
  bf16* Wih1b = (bf16*)wsalloc(2ull * 1536 * 1024 * 2);
  bf16* Whh1b = (bf16*)wsalloc(2ull * 1536 * 512 * 2);
  bf16* attnWb = (bf16*)wsalloc(1024ull * 1024 * 2);
  bf16* fcWb = (bf16*)wsalloc(10ull * 1024 * 2);
  bf16* xb = (bf16*)wsalloc(M * 128 * 2);
  float* xg = (float*)wsalloc(2ull * M * 1536 * 4);
  bf16* act0b = (bf16*)wsalloc(M * 1024 * 2);
  bf16* act1b = (bf16*)wsalloc(M * 1024 * 2);
  float* act1f = (float*)wsalloc(M * 1024 * 4);

  auto conv = [&](const float* src, bf16* dst, size_t n) {
    f32_to_bf16_kernel<<<dim3((unsigned)((n + 255) / 256)), dim3(256), 0,
                         stream>>>(src, dst, (int)n);
  };
  conv(Wih0, Wih0b, 2ull * 1536 * 128);
  conv(Whh0, Whh0b, 2ull * 1536 * 512);
  conv(Wih1, Wih1b, 2ull * 1536 * 1024);
  conv(Whh1, Whh1b, 2ull * 1536 * 512);
  conv(attnW, attnWb, 1024ull * 1024);
  conv(fcW, fcWb, 10ull * 1024);
  conv(x, xb, M * 128);

  for (int d = 0; d < 2; d++) {
    gemm_wmma_bf16<0><<<dim3(1536 / 64, (unsigned)(M / 32)), 256, 0, stream>>>(
        xb, Wih0b + (size_t)d * 1536 * 128, bih0 + d * 1536,
        xg + (size_t)d * M * 1536, nullptr, nullptr, (int)M, 1536, 128, 1536);
  }
  gru_layer_kernel<<<dim3(4), dim3(1024), 0, stream>>>(xg, Whh0b, bhh0, act0b,
                                                       nullptr, 0);
  for (int d = 0; d < 2; d++) {
    gemm_wmma_bf16<0><<<dim3(1536 / 64, (unsigned)(M / 32)), 256, 0, stream>>>(
        act0b, Wih1b + (size_t)d * 1536 * 1024, bih1 + d * 1536,
        xg + (size_t)d * M * 1536, nullptr, nullptr, (int)M, 1536, 1024, 1536);
  }
  gru_layer_kernel<<<dim3(4), dim3(1024), 0, stream>>>(xg, Whh1b, bhh1, act1b,
                                                       act1f, 1);
  gemm_wmma_bf16<1><<<dim3(1024 / 64, (unsigned)(M / 32)), 256, 0, stream>>>(
      act1b, attnWb, attnb, nullptr, act0b, act1f, (int)M, 1024, 1024, 1024);
  gemm_wmma_bf16<0><<<dim3(1, (unsigned)(M / 32)), 256, 0, stream>>>(
      act0b, fcWb, fcb, out, nullptr, nullptr, (int)M, 10, 1024, 10);
}
